// TransformerBlock_43267500540230
// MI455X (gfx1250) — matrix operand order repaired and checked by reading, not compiled, not run on hardware
//
#include <hip/hip_runtime.h>
#include <hip/hip_bf16.h>
#include <math.h>

#define B_   4
#define L_   1024
#define C_   1024
#define H_   16
#define D_   64
#define HID_ 4096
#define EPS_ 1e-6f

typedef __attribute__((ext_vector_type(16))) __bf16 v16bf;
typedef __attribute__((ext_vector_type(8)))  __bf16 v8bf;
typedef __attribute__((ext_vector_type(8)))  float  v8f;

__device__ __forceinline__ unsigned short f32_to_bf16(float f) {
  unsigned int u = __float_as_uint(f);
  u += 0x7fffu + ((u >> 16) & 1u);          // round-to-nearest-even
  return (unsigned short)(u >> 16);
}

// --------- async global->LDS 16-byte copy (ASYNCcnt-tracked on gfx1250) ------
__device__ __forceinline__ void lds_copy16(const unsigned short* g,
                                           unsigned short* l) {
#if defined(__gfx1250__)
  // LDS byte offset = low 32 bits of the generic (shared-aperture) address.
  unsigned lo = (unsigned)(unsigned long long)l;
  asm volatile("global_load_async_to_lds_b128 %0, %1, off"
               :: "v"(lo), "v"((unsigned long long)g)
               : "memory");
#else
  *(uint4*)l = *(const uint4*)g;
#endif
}

__device__ __forceinline__ void lds_copy_wait() {
#if defined(__gfx1250__)
  asm volatile("s_wait_asynccnt 0" ::: "memory");
#endif
}

// ---------------------------------------------------------------- weight cast
__global__ void cast_bf16_kernel(const float* __restrict__ in,
                                 unsigned short* __restrict__ out, int n) {
  int stride = gridDim.x * blockDim.x;
  for (int i = blockIdx.x * blockDim.x + threadIdx.x; i < n; i += stride)
    out[i] = f32_to_bf16(in[i]);
}

// --------------------------------------------------- mod = silu(style)@adaW^T
__global__ __launch_bounds__(256) void mod_kernel(
    const float* __restrict__ style, const float* __restrict__ ada_w,
    const float* __restrict__ ada_b, float* __restrict__ mod) {
  int idx = blockIdx.x * 256 + threadIdx.x;   // B*4C = 16384
  int bb  = idx >> 12;
  int j   = idx & 4095;
  const float* srow = style + bb * C_;
  const float* wrow = ada_w + (size_t)j * C_;
  float acc = ada_b[j];
  for (int k = 0; k < C_; ++k) {
    float s = srow[k];
    s = s / (1.f + __expf(-s));
    acc += s * wrow[k];
  }
  mod[idx] = acc;
}

// ----------------------------------------- fused LayerNorm + AdaLN modulation
__global__ __launch_bounds__(256) void adaln_kernel(
    const float* __restrict__ x, const float* __restrict__ g,
    const float* __restrict__ bta, const float* __restrict__ mod,
    int sc_off, int sh_off, unsigned short* __restrict__ out) {
  __shared__ float red1[256], red2[256];
  const int row = blockIdx.x;
  const int bb  = row >> 10;                  // / L_
  const float* xr = x + (size_t)row * C_;
  const int tid = threadIdx.x;
  float s1 = 0.f, s2 = 0.f, vals[4];
#pragma unroll
  for (int i = 0; i < 4; ++i) {
    float v = xr[tid + i * 256];
    vals[i] = v; s1 += v; s2 += v * v;
  }
  red1[tid] = s1; red2[tid] = s2;
  __syncthreads();
  for (int s = 128; s > 0; s >>= 1) {
    if (tid < s) { red1[tid] += red1[tid + s]; red2[tid] += red2[tid + s]; }
    __syncthreads();
  }
  float mean = red1[0] * (1.f / C_);
  float var  = red2[0] * (1.f / C_) - mean * mean;
  float inv  = rsqrtf(var + EPS_);
  const float* sc = mod + (size_t)bb * (4 * C_) + sc_off;
  const float* sh = mod + (size_t)bb * (4 * C_) + sh_off;
  unsigned short* orow = out + (size_t)row * C_;
#pragma unroll
  for (int i = 0; i < 4; ++i) {
    int c = tid + i * 256;
    float y = (vals[i] - mean) * inv * g[c] + bta[c];
    y = y * (1.f + sc[c]) + sh[c];
    orow[c] = f32_to_bf16(y);
  }
}

// ------------------------------------------------- generic bf16 WMMA GEMM
// D[m,n] = sum_k A[m,k]*W[n,k] (+bias) (+residual) (+gelu), out f32 and/or bf16
// A block-panel (32 rows x 64 K) is double-buffered in LDS via async copies.
__global__ __launch_bounds__(256) void gemm_bf16_kernel(
    const unsigned short* __restrict__ A,      // M x K bf16
    const unsigned short* __restrict__ W,      // N x K bf16 (i.e. W^T row-major)
    const float* __restrict__ bias,            // N or null
    const float* __restrict__ resid,           // M x N or null
    float* __restrict__ outf,                  // M x N or null
    unsigned short* __restrict__ outb,         // M x N or null
    int M, int N, int K, int act) {
  __shared__ __align__(16) unsigned short sA[2][32 * 64];  // 2 x 4KB
  const int tid  = threadIdx.x;
  const int wave = tid >> 5;                   // wave32
  const int lane = tid & 31;
  const int r    = lane & 15;
  const int hl   = lane >> 4;
  const int mt   = wave & 1;
  const int m0b  = blockIdx.x * 32;
  const int n0   = blockIdx.y * 256 + (wave >> 1) * 64;

  // cooperative staging map: 256 threads x 16B = one 32x64 bf16 panel
  const int srow = tid >> 3, sch = tid & 7;
  const unsigned short* gA = A + (size_t)(m0b + srow) * K + sch * 8;
  unsigned short* lA[2] = { &sA[0][srow * 64 + sch * 8],
                            &sA[1][srow * 64 + sch * 8] };

  lds_copy16(gA, lA[0]);                       // preload panel 0
  v8f acc[4] = {};
  const int np = K >> 6;                       // 64-wide K panels
  for (int p = 0; p < np; ++p) {
    lds_copy_wait();
    __syncthreads();                           // panel p resident & visible
    if (p + 1 < np)
      lds_copy16(gA + (size_t)(p + 1) * 64, lA[(p + 1) & 1]);  // overlap copy
    const unsigned short* Abuf = sA[p & 1];
    const int kp = p << 6;
#pragma unroll
    for (int kk = 0; kk < 64; kk += 32) {
      union { v16bf v; v8bf h2[2]; } af;
      const unsigned short* Ar = Abuf + (mt * 16 + r) * 64 + kk + hl * 16;
      af.h2[0] = *(const v8bf*)(Ar);           // ds_load_b128
      af.h2[1] = *(const v8bf*)(Ar + 8);
#pragma unroll
      for (int s = 0; s < 4; ++s) {
        const unsigned short* Wrow =
            W + (size_t)(n0 + s * 16 + r) * K + kp + kk + hl * 16;
        v16bf bfr = *(const v16bf*)Wrow;
        acc[s] = __builtin_amdgcn_wmma_f32_16x16x32_bf16(
            false, af.v, false, bfr, (short)0, acc[s], false, false);
      }
    }
  }

  const int m0 = m0b + mt * 16;
#pragma unroll
  for (int s = 0; s < 4; ++s) {
    int col = n0 + s * 16 + r;
    float bv = bias ? bias[col] : 0.f;
#pragma unroll
    for (int e = 0; e < 8; ++e) {
      int row = m0 + e + hl * 8;               // D layout: M = e + 8*half
      float v = acc[s][e] + bv;
      size_t idx = (size_t)row * N + col;
      if (resid) v += resid[idx];
      if (act == 1) v = 0.5f * v * (1.f + erff(v * 0.70710678118654752440f));
      if (outf) outf[idx] = v;
      if (outb) outb[idx] = f32_to_bf16(v);
    }
  }
}

// ------------------- RoPE + pack Q,K head-major bf16, V transposed (D x L)
__global__ __launch_bounds__(256) void rope_kernel(
    const float* __restrict__ qkv, const float* __restrict__ freqs,
    unsigned short* __restrict__ Qb, unsigned short* __restrict__ Kb,
    unsigned short* __restrict__ Vt) {
  int idx = blockIdx.x * 256 + threadIdx.x;    // ((b*H+h)*L + l)*32 + dp
  int dp = idx & 31;
  int t  = idx >> 5;
  int l  = t & (L_ - 1);
  int t2 = t >> 10;
  int hh = t2 & (H_ - 1);
  int b  = t2 >> 4;
  float c = freqs[(l * 32 + dp) * 2 + 0];
  float s = freqs[(l * 32 + dp) * 2 + 1];
  size_t base = ((size_t)(b * L_ + l)) * (3 * C_) + hh * D_ + dp * 2;
  float q0 = qkv[base],          q1 = qkv[base + 1];
  float k0 = qkv[base + C_],     k1 = qkv[base + C_ + 1];
  float v0 = qkv[base + 2 * C_], v1 = qkv[base + 2 * C_ + 1];
  size_t ob = ((size_t)(b * H_ + hh) * L_ + l) * D_ + dp * 2;
  Qb[ob]     = f32_to_bf16(q0 * c - q1 * s);
  Qb[ob + 1] = f32_to_bf16(q1 * c + q0 * s);
  Kb[ob]     = f32_to_bf16(k0 * c - k1 * s);
  Kb[ob + 1] = f32_to_bf16(k1 * c + k0 * s);
  size_t vb = ((size_t)(b * H_ + hh) * D_ + dp * 2) * L_ + l;
  Vt[vb]      = f32_to_bf16(v0);
  Vt[vb + L_] = f32_to_bf16(v1);
}

// ------------- attention: one block per (b,h, 16-query strip), WMMA both ways
__global__ __launch_bounds__(256) void attn_kernel(
    const unsigned short* __restrict__ Qb, const unsigned short* __restrict__ Kb,
    const unsigned short* __restrict__ Vt, const float* __restrict__ bias,
    unsigned short* __restrict__ out_b) {
  extern __shared__ char smem[];
  float*          sS   = (float*)smem;                              // 16x1024 f32
  unsigned short* sP   = (unsigned short*)(smem + 16 * L_ * 4);     // 16x1024 bf16
  float*          sRed = (float*)(smem + 16 * L_ * 4 + 16 * L_ * 2);// 256
  float*          sMax = sRed + 256;                                // 16
  float*          sSum = sMax + 16;                                 // 16

  const int tid  = threadIdx.x;
  const int wave = tid >> 5;
  const int lane = tid & 31;
  const int r    = lane & 15;
  const int hl   = lane >> 4;
  const int bh = blockIdx.x >> 6;              // 64 q-tiles per (b,h)
  const int qt = blockIdx.x & 63;
  const int q0 = qt * 16;
  const int h  = bh & (H_ - 1);

  // phase 1: S = Q K^T / 8 + bias  -> LDS strip (wave owns 128 columns)
  const unsigned short* Qrow = Qb + (size_t)(bh * L_ + q0 + r) * D_;
  const int c0 = wave * 128;
  v8f acc[8] = {};
#pragma unroll
  for (int kk = 0; kk < D_; kk += 32) {
    union { v16bf v; v8bf h2[2]; } af;
    af.h2[0] = *(const v8bf*)(Qrow + kk + hl * 16);
    af.h2[1] = *(const v8bf*)(Qrow + kk + 8 + hl * 16);
#pragma unroll
    for (int s = 0; s < 8; ++s) {
      const unsigned short* Krow =
          Kb + (size_t)(bh * L_ + c0 + s * 16 + r) * D_ + kk + hl * 16;
      v16bf bfr = *(const v16bf*)Krow;
      acc[s] = __builtin_amdgcn_wmma_f32_16x16x32_bf16(
          false, af.v, false, bfr, (short)0, acc[s], false, false);
    }
  }
#pragma unroll
  for (int s = 0; s < 8; ++s) {
    int col = c0 + s * 16 + r;
#pragma unroll
    for (int e = 0; e < 8; ++e) {
      int qr = q0 + e + hl * 8;
      float sc = acc[s][e] * 0.125f + bias[((size_t)h * L_ + qr) * L_ + col];
      sS[(e + hl * 8) * L_ + col] = sc;
    }
  }
  __syncthreads();

  // phase 2: softmax over 1024 keys, P -> bf16 in LDS
  {
    int row = tid >> 4, sub = tid & 15;
    float mx = -3.4e38f;
    for (int j = sub; j < L_; j += 16) mx = fmaxf(mx, sS[row * L_ + j]);
    sRed[tid] = mx;
    __syncthreads();
    if (sub == 0) {
      float m = sRed[tid];
      for (int t = 1; t < 16; ++t) m = fmaxf(m, sRed[tid + t]);
      sMax[row] = m;
    }
    __syncthreads();
    float m = sMax[row], sum = 0.f;
    for (int j = sub; j < L_; j += 16) {
      float p = __expf(sS[row * L_ + j] - m);
      sum += p;
      sP[row * L_ + j] = f32_to_bf16(p);
    }
    sRed[tid] = sum;
    __syncthreads();
    if (sub == 0) {
      float s2 = 0.f;
      for (int t = 0; t < 16; ++t) s2 += sRed[tid + t];
      sSum[row] = s2;
    }
    __syncthreads();
  }

  // phase 3: O = P V  (waves 0..3, one 16x16 d-tile each, K = 1024)
  if (wave < 4) {
    const int d0 = wave * 16;
    v8f oacc = {};
    const unsigned short* Prow = sP + r * L_;
    const unsigned short* Vrow = Vt + (size_t)(bh * D_ + d0 + r) * L_;
    for (int k = 0; k < L_; k += 32) {
      union { v16bf v; v8bf h2[2]; } af;
      af.h2[0] = *(const v8bf*)(Prow + k + hl * 16);          // ds_load_b128
      af.h2[1] = *(const v8bf*)(Prow + k + 8 + hl * 16);
      v16bf bfr = *(const v16bf*)(Vrow + k + hl * 16);
      oacc = __builtin_amdgcn_wmma_f32_16x16x32_bf16(
          false, af.v, false, bfr, (short)0, oacc, false, false);
    }
    const int b = bh >> 4;
#pragma unroll
    for (int e = 0; e < 8; ++e) {
      int qr = q0 + e + hl * 8;
      float val = oacc[e] / sSum[e + hl * 8];
      out_b[(size_t)(b * L_ + qr) * C_ + h * D_ + d0 + r] = f32_to_bf16(val);
    }
  }
}

// ---------------------------------------------------------------- launcher
extern "C" void kernel_launch(void* const* d_in, const int* in_sizes, int n_in,
                              void* d_out, int out_size, void* d_ws, size_t ws_size,
                              hipStream_t stream) {
  (void)in_sizes; (void)n_in; (void)out_size; (void)ws_size;
  const float* x      = (const float*)d_in[0];
  const float* style  = (const float*)d_in[1];
  const float* freqs  = (const float*)d_in[2];
  const float* abias  = (const float*)d_in[3];
  const float* ln1_g  = (const float*)d_in[4];
  const float* ln1_b  = (const float*)d_in[5];
  const float* ln2_g  = (const float*)d_in[6];
  const float* ln2_b  = (const float*)d_in[7];
  const float* qkv_w  = (const float*)d_in[8];
  const float* proj_w = (const float*)d_in[9];
  const float* proj_b = (const float*)d_in[10];
  const float* fc1_w  = (const float*)d_in[11];
  const float* fc1_b  = (const float*)d_in[12];
  const float* fc2_w  = (const float*)d_in[13];
  const float* fc2_b  = (const float*)d_in[14];
  const float* ada_w  = (const float*)d_in[15];
  const float* ada_b  = (const float*)d_in[16];
  float* out = (float*)d_out;

  char* ws = (char*)d_ws;
  size_t off = 0;
  auto carve = [&](size_t bytes) -> char* {
    char* p = ws + off;
    off += (bytes + 255) & ~(size_t)255;
    return p;
  };
  float*          modf   = (float*)carve((size_t)B_ * 4 * C_ * 4);
  unsigned short* qkvwb  = (unsigned short*)carve((size_t)3 * C_ * C_ * 2);
  unsigned short* projwb = (unsigned short*)carve((size_t)C_ * C_ * 2);
  unsigned short* fc1wb  = (unsigned short*)carve((size_t)HID_ * C_ * 2);
  unsigned short* fc2wb  = (unsigned short*)carve((size_t)C_ * HID_ * 2);
  unsigned short* h1b    = (unsigned short*)carve((size_t)B_ * L_ * C_ * 2);
  float*          qkvf   = (float*)carve((size_t)B_ * L_ * 3 * C_ * 4);
  unsigned short* Qb     = (unsigned short*)carve((size_t)B_ * L_ * C_ * 2);
  unsigned short* Kb     = (unsigned short*)carve((size_t)B_ * L_ * C_ * 2);
  unsigned short* Vt     = (unsigned short*)carve((size_t)B_ * L_ * C_ * 2);
  unsigned short* aob    = (unsigned short*)carve((size_t)B_ * L_ * C_ * 2);
  float*          x1f    = (float*)carve((size_t)B_ * L_ * C_ * 4);
  unsigned short* h3b = (unsigned short*)qkvf;  // reuse: qkv dead after rope
  unsigned short* h2b = Qb;                     // reuse: Qb dead after attn

  const int TPB = 256;
  cast_bf16_kernel<<<2048, TPB, 0, stream>>>(qkv_w,  qkvwb,  3 * C_ * C_);
  cast_bf16_kernel<<<2048, TPB, 0, stream>>>(proj_w, projwb, C_ * C_);
  cast_bf16_kernel<<<2048, TPB, 0, stream>>>(fc1_w,  fc1wb,  HID_ * C_);
  cast_bf16_kernel<<<2048, TPB, 0, stream>>>(fc2_w,  fc2wb,  C_ * HID_);

  mod_kernel<<<(B_ * 4 * C_) / TPB, TPB, 0, stream>>>(style, ada_w, ada_b, modf);

  adaln_kernel<<<B_ * L_, TPB, 0, stream>>>(x, ln1_g, ln1_b, modf, 0, C_, h1b);

  gemm_bf16_kernel<<<dim3((B_ * L_) / 32, (3 * C_) / 256), TPB, 0, stream>>>(
      h1b, qkvwb, nullptr, nullptr, qkvf, nullptr, B_ * L_, 3 * C_, C_, 0);

  rope_kernel<<<(B_ * H_ * L_ * (D_ / 2)) / TPB, TPB, 0, stream>>>(
      qkvf, freqs, Qb, Kb, Vt);

  size_t shmem = (size_t)16 * L_ * 4 + 16 * L_ * 2 + 256 * 4 + 32 * 4;
  attn_kernel<<<B_ * H_ * (L_ / 16), TPB, shmem, stream>>>(Qb, Kb, Vt, abias, aob);

  gemm_bf16_kernel<<<dim3((B_ * L_) / 32, C_ / 256), TPB, 0, stream>>>(
      aob, projwb, proj_b, x, x1f, nullptr, B_ * L_, C_, C_, 0);

  adaln_kernel<<<B_ * L_, TPB, 0, stream>>>(x1f, ln2_g, ln2_b, modf, 2 * C_, 3 * C_, h2b);

  gemm_bf16_kernel<<<dim3((B_ * L_) / 32, HID_ / 256), TPB, 0, stream>>>(
      h2b, fc1wb, fc1_b, nullptr, nullptr, h3b, B_ * L_, HID_, C_, 1);

  gemm_bf16_kernel<<<dim3((B_ * L_) / 32, C_ / 256), TPB, 0, stream>>>(
      h3b, fc2wb, fc2_b, x1f, out, nullptr, B_ * L_, C_, HID_, 0);
}
